// ASAPooling_76501957477035
// MI455X (gfx1250) — matrix operand order repaired and checked by reading, compiled, not run on hardware
//
#include <hip/hip_runtime.h>
#include <hip/hip_bf16.h>

// ASAPooling for MI455X (gfx1250).
// Sparse/scatter phases in f32 scalar code (memory/atomic bound);
// dense coarsening GEMM A_c = Sp^T @ T via v_wmma_f32_16x16x32_bf16,
// 2x2 register-blocked (32x32 output per wave) to halve loads per WMMA.

typedef __attribute__((ext_vector_type(16))) __bf16 v16bf;
typedef __attribute__((ext_vector_type(8)))  __bf16 v8bf;
typedef __attribute__((ext_vector_type(8)))  float  v8f;

#define NEG_SLOPE 0.2f

__device__ inline __bf16 f2bf(float f) {
  unsigned u = __float_as_uint(f);
  unsigned r = (u + 0x7FFFu + ((u >> 16) & 1u)) >> 16;  // round-to-nearest-even
  unsigned short h = (unsigned short)r;
  return __builtin_bit_cast(__bf16, h);
}

__device__ inline void atomic_max_f32(float* addr, float val) {
  // int/uint monotonic trick; addr initialized to -inf
  if (val >= 0.f) atomicMax((int*)addr, __float_as_int(val));
  else            atomicMin((unsigned int*)addr, __float_as_uint(val));
}

__device__ inline float wave_sum(float v) {
#pragma unroll
  for (int o = 16; o > 0; o >>= 1) v += __shfl_xor(v, o, 32);
  return v;
}

__device__ inline int edge_src(const int* s, long e, long E0) { return e < E0 ? s[e] : (int)(e - E0); }
__device__ inline int edge_dst(const int* d, long e, long E0) { return e < E0 ? d[e] : (int)(e - E0); }

// ---------------- fills ----------------
__global__ void k_fill_f32(float* __restrict__ p, float v, long n) {
  long i = (long)blockIdx.x * blockDim.x + threadIdx.x;
  long st = (long)gridDim.x * blockDim.x;
  for (; i < n; i += st) p[i] = v;
}
__global__ void k_fill_i32(int* __restrict__ p, int v, long n) {
  long i = (long)blockIdx.x * blockDim.x + threadIdx.x;
  long st = (long)gridDim.x * blockDim.x;
  for (; i < n; i += st) p[i] = v;
}

// v = lin_w @ att_w[:D]; c0 = lin_b . att_w[:D] + att_b
__global__ void k_vvec(const float* __restrict__ lin_w, const float* __restrict__ lin_b,
                       const float* __restrict__ att_w, const float* __restrict__ att_b,
                       float* __restrict__ vvec, float* __restrict__ c0, int D) {
  int i = threadIdx.x;
  if (i < D) {
    float s = 0.f;
    for (int o = 0; o < D; ++o) s += lin_w[(long)i * D + o] * att_w[o];
    vvec[i] = s;
  }
  if (i == 0) {
    float s = att_b[0];
    for (int o = 0; o < D; ++o) s += lin_b[o] * att_w[o];
    c0[0] = s;
  }
}

// x_q[dst, d] = segment_max over edges of x[src, d]
__global__ void k_xq_max(const float* __restrict__ x, const int* __restrict__ srcI,
                         const int* __restrict__ dstI, float* __restrict__ xq,
                         long E0, long Et, int D) {
  long total = Et * D;
  long i = (long)blockIdx.x * blockDim.x + threadIdx.x;
  long st = (long)gridDim.x * blockDim.x;
  for (; i < total; i += st) {
    long e = i / D; int d = (int)(i - e * D);
    int s = edge_src(srcI, e, E0);
    int dd = edge_dst(dstI, e, E0);
    atomic_max_f32(&xq[(long)dd * D + d], x[(long)s * D + d]);
  }
}

// q[n] = xq[n]·v + c0 ; p[n] = x[n]·att_w[D:]
__global__ void k_node_qp(const float* __restrict__ xq, const float* __restrict__ x,
                          const float* __restrict__ vvec, const float* __restrict__ c0,
                          const float* __restrict__ att_w,
                          float* __restrict__ qv, float* __restrict__ pv, int N, int D) {
  int wid = (blockIdx.x * blockDim.x + threadIdx.x) >> 5;
  int lane = threadIdx.x & 31;
  if (wid >= N) return;
  float q = 0.f, p = 0.f;
  for (int d = lane; d < D; d += 32) {
    q += xq[(long)wid * D + d] * vvec[d];
    p += x[(long)wid * D + d] * att_w[D + d];
  }
  q = wave_sum(q); p = wave_sum(p);
  if (lane == 0) { qv[wid] = q + c0[0]; pv[wid] = p; }
}

__global__ void k_score1(const int* __restrict__ srcI, const int* __restrict__ dstI,
                         const float* __restrict__ qv, const float* __restrict__ pv,
                         float* __restrict__ score, float* __restrict__ smax,
                         long E0, long Et) {
  long e = (long)blockIdx.x * blockDim.x + threadIdx.x;
  long st = (long)gridDim.x * blockDim.x;
  for (; e < Et; e += st) {
    int s = edge_src(srcI, e, E0);
    int dd = edge_dst(dstI, e, E0);
    float sc = qv[dd] + pv[s];
    sc = sc > 0.f ? sc : NEG_SLOPE * sc;
    score[e] = sc;
    atomic_max_f32(&smax[dd], sc);
  }
}

__global__ void k_score2(const int* __restrict__ dstI, float* __restrict__ score,
                         const float* __restrict__ smax, float* __restrict__ denom,
                         long E0, long Et) {
  long e = (long)blockIdx.x * blockDim.x + threadIdx.x;
  long st = (long)gridDim.x * blockDim.x;
  for (; e < Et; e += st) {
    int dd = edge_dst(dstI, e, E0);
    float ex = __expf(score[e] - smax[dd]);
    score[e] = ex;
    atomicAdd(&denom[dd], ex);
  }
}

__global__ void k_alpha(const int* __restrict__ dstI, const float* __restrict__ score,
                        const float* __restrict__ denom, float* __restrict__ alphav,
                        long E0, long Et) {
  long e = (long)blockIdx.x * blockDim.x + threadIdx.x;
  long st = (long)gridDim.x * blockDim.x;
  for (; e < Et; e += st) {
    int dd = edge_dst(dstI, e, E0);
    alphav[e] = score[e] / denom[dd];
  }
}

// xn[dst, :] += alpha_e * x[src, :]
__global__ void k_xn(const int* __restrict__ srcI, const int* __restrict__ dstI,
                     const float* __restrict__ alphav, const float* __restrict__ x,
                     float* __restrict__ xn, long E0, long Et, int D) {
  long total = Et * D;
  long i = (long)blockIdx.x * blockDim.x + threadIdx.x;
  long st = (long)gridDim.x * blockDim.x;
  for (; i < total; i += st) {
    long e = i / D; int d = (int)(i - e * D);
    int s = edge_src(srcI, e, E0);
    int dd = edge_dst(dstI, e, E0);
    atomicAdd(&xn[(long)dd * D + d], alphav[e] * x[(long)s * D + d]);
  }
}

// LEConv node scalars
__global__ void k_le(const float* __restrict__ xn,
                     const float* __restrict__ le1_w, const float* __restrict__ le1_b,
                     const float* __restrict__ le2_w, const float* __restrict__ le3_w,
                     float* __restrict__ av, float* __restrict__ bv, float* __restrict__ c3v,
                     int N, int D) {
  int wid = (blockIdx.x * blockDim.x + threadIdx.x) >> 5;
  int lane = threadIdx.x & 31;
  if (wid >= N) return;
  float s1 = 0.f, s2 = 0.f, s3 = 0.f;
  for (int d = lane; d < D; d += 32) {
    float v = xn[(long)wid * D + d];
    s1 += v * le1_w[d]; s2 += v * le2_w[d]; s3 += v * le3_w[d];
  }
  s1 = wave_sum(s1); s2 = wave_sum(s2); s3 = wave_sum(s3);
  if (lane == 0) { av[wid] = s1 + le1_b[0]; bv[wid] = s2; c3v[wid] = s3; }
}

__global__ void k_agg(const int* __restrict__ srcI, const int* __restrict__ dstI,
                      const float* __restrict__ av, const float* __restrict__ bv,
                      float* __restrict__ aggv, long E0, long Et) {
  long e = (long)blockIdx.x * blockDim.x + threadIdx.x;
  long st = (long)gridDim.x * blockDim.x;
  for (; e < Et; e += st) {
    int s = edge_src(srcI, e, E0);
    int dd = edge_dst(dstI, e, E0);
    atomicAdd(&aggv[dd], av[s] - bv[dd]);
  }
}

__global__ void k_fit(const float* __restrict__ aggv, const float* __restrict__ c3v,
                      const float* __restrict__ le3_b, float* __restrict__ fitv, int N) {
  int i = blockIdx.x * blockDim.x + threadIdx.x;
  if (i >= N) return;
  float z = aggv[i] + c3v[i] + le3_b[0];
  fitv[i] = 1.f / (1.f + __expf(-z));
}

// exact top-k via rank (desc value, asc index tiebreak) -- matches jax.lax.top_k order
__global__ void k_topk(const float* __restrict__ fitv, int* __restrict__ perm,
                       int* __restrict__ invperm, float* __restrict__ outPerm,
                       int N, int Kc) {
  int i = blockIdx.x * blockDim.x + threadIdx.x;
  if (i >= N) return;
  float fi = fitv[i];
  int rank = 0;
  for (int j = 0; j < N; ++j) {
    float fj = fitv[j];
    rank += (fj > fi) || (fj == fi && j < i);
  }
  if (rank < Kc) {
    perm[rank] = i;
    invperm[i] = rank;
    outPerm[rank] = (float)i;
  }
}

__global__ void k_xout(const int* __restrict__ perm, const float* __restrict__ fitv,
                       const float* __restrict__ xn, float* __restrict__ outX,
                       int Kc, int D) {
  long total = (long)Kc * D;
  long i = (long)blockIdx.x * blockDim.x + threadIdx.x;
  long st = (long)gridDim.x * blockDim.x;
  for (; i < total; i += st) {
    int r = (int)(i / D); int d = (int)(i - (long)r * D);
    int n = perm[r];
    outX[i] = xn[(long)n * D + d] * fitv[n];
  }
}

// Sp[src, invperm[dst]] += alpha_e  (dense [N, Kc] f32)
__global__ void k_sp(const int* __restrict__ srcI, const int* __restrict__ dstI,
                     const int* __restrict__ invperm, const float* __restrict__ alphav,
                     float* __restrict__ Sp, long E0, long Et, int Kc) {
  long e = (long)blockIdx.x * blockDim.x + threadIdx.x;
  long st = (long)gridDim.x * blockDim.x;
  for (; e < Et; e += st) {
    int s = edge_src(srcI, e, E0);
    int dd = edge_dst(dstI, e, E0);
    int j = invperm[dd];
    if (j >= 0) atomicAdd(&Sp[(long)s * Kc + j], alphav[e]);
  }
}

// T[src, :] += w_e * Sp[dst, :]   (sparse A times Sp; skip zero entries)
__global__ void k_tscatter(const int* __restrict__ srcI, const int* __restrict__ dstI,
                           const float* __restrict__ ew, const float* __restrict__ Sp,
                           float* __restrict__ T, long E0, long Et, int Kc) {
  long total = Et * Kc;
  long i = (long)blockIdx.x * blockDim.x + threadIdx.x;
  long st = (long)gridDim.x * blockDim.x;
  for (; i < total; i += st) {
    long e = i / Kc; int j = (int)(i - e * Kc);
    int dd = edge_dst(dstI, e, E0);
    float v = Sp[(long)dd * Kc + j];
    if (v != 0.f) {
      int s = edge_src(srcI, e, E0);
      float w = e < E0 ? ew[e] : 1.f;
      atomicAdd(&T[(long)s * Kc + j], w * v);
    }
  }
}

// dst[m*N + n] = bf16(src[n*Kc + m])  : transpose f32 [N,Kc] -> bf16 [Kc,N] (K-major)
__global__ void k_transpose_bf16(const float* __restrict__ src, __bf16* __restrict__ dst,
                                 int N, int Kc) {
  long total = (long)Kc * N;
  long i = (long)blockIdx.x * blockDim.x + threadIdx.x;
  long st = (long)gridDim.x * blockDim.x;
  for (; i < total; i += st) {
    long m = i / N; int n = (int)(i - m * N);
    dst[i] = f2bf(src[(long)n * Kc + m]);
  }
}

// ---- WMMA fragment loads (layouts per CDNA5 ISA 7.12.2) ----
// A 16x32 bf16: lanes 0-15 hold K {0..7,16..23}; lanes 16-31 hold K {8..15,24..31}
__device__ inline v16bf load_frag_a(const __bf16* __restrict__ row, int k0, int half) {
  v8bf lo = *(const v8bf*)(row + k0 + half * 8);
  v8bf hi = *(const v8bf*)(row + k0 + 16 + half * 8);
  v16bf a;
#pragma unroll
  for (int i = 0; i < 8; ++i) { a[i] = lo[i]; a[8 + i] = hi[i]; }
  return a;
}
// B 32x16 bf16 (from B^T rows): lanes 0-15 hold K {0..7,16..23}; lanes 16-31 hold K {8..15,24..31}
__device__ inline v16bf load_frag_b(const __bf16* __restrict__ row, int k0, int half) {
  v8bf lo = *(const v8bf*)(row + k0 + half * 8);
  v8bf hi = *(const v8bf*)(row + k0 + 16 + half * 8);
  v16bf b;
#pragma unroll
  for (int i = 0; i < 8; ++i) { b[i] = lo[i]; b[8 + i] = hi[i]; }
  return b;
}

// C[M,Nn] = A[M,K](bf16, K-major) @ B with Bt[Nn,K](bf16, K-major) holding B^T.
// One wave per 32x32 output block (2x2 tiles of 16x16): each A/B fragment feeds
// two WMMAs, halving L2 load traffic per matrix op.
__global__ void k_gemm_bf16_2x2(const __bf16* __restrict__ A, const __bf16* __restrict__ Bt,
                                float* __restrict__ C, int M, int Nn, int K, int ldc,
                                int totalBlocks) {
  int wid = (blockIdx.x * blockDim.x + threadIdx.x) >> 5;
  int lane = threadIdx.x & 31;
  if (wid >= totalBlocks) return;
  int blocksN = Nn >> 5;
  int bm = wid / blocksN, bn = wid - bm * blocksN;
  int half = lane >> 4;       // 0: lanes 0-15, 1: lanes 16-31
  int l = lane & 15;
  const __bf16* arow0 = A + (long)(bm * 32 + l) * K;        // rows r..r+15
  const __bf16* arow1 = arow0 + (long)16 * K;               // rows r+16..r+31
  const __bf16* brow0 = Bt + (long)(bn * 32 + l) * K;       // cols c..c+15
  const __bf16* brow1 = brow0 + (long)16 * K;               // cols c+16..c+31
  v8f acc00 = {}, acc01 = {}, acc10 = {}, acc11 = {};
  for (int k0 = 0; k0 < K; k0 += 32) {
    v16bf a0 = load_frag_a(arow0, k0, half);
    v16bf a1 = load_frag_a(arow1, k0, half);
    v16bf b0 = load_frag_b(brow0, k0, half);
    v16bf b1 = load_frag_b(brow1, k0, half);
    acc00 = __builtin_amdgcn_wmma_f32_16x16x32_bf16(false, a0, false, b0, (short)0, acc00, false, false);
    acc01 = __builtin_amdgcn_wmma_f32_16x16x32_bf16(false, a0, false, b1, (short)0, acc01, false, false);
    acc10 = __builtin_amdgcn_wmma_f32_16x16x32_bf16(false, a1, false, b0, (short)0, acc10, false, false);
    acc11 = __builtin_amdgcn_wmma_f32_16x16x32_bf16(false, a1, false, b1, (short)0, acc11, false, false);
  }
  // C/D layout: lanes 0-15 -> rows base+r, lanes 16-31 -> rows base+8+r; col = base+l
  int col0 = bn * 32 + l, col1 = col0 + 16;
  int rb0 = bm * 32 + half * 8, rb1 = rb0 + 16;
#pragma unroll
  for (int r = 0; r < 8; ++r) {
    C[(long)(rb0 + r) * ldc + col0] = acc00[r];
    C[(long)(rb0 + r) * ldc + col1] = acc01[r];
    C[(long)(rb1 + r) * ldc + col0] = acc10[r];
    C[(long)(rb1 + r) * ldc + col1] = acc11[r];
  }
}

__global__ void k_diag0(float* __restrict__ Ac, int Kc) {
  int i = blockIdx.x * blockDim.x + threadIdx.x;
  if (i < Kc) Ac[(long)i * Kc + i] = 0.f;
}

// ---------------------------------------------------------------------------

extern "C" void kernel_launch(void* const* d_in, const int* in_sizes, int n_in,
                              void* d_out, int out_size, void* d_ws, size_t ws_size,
                              hipStream_t stream) {
  const float* x     = (const float*)d_in[0];
  const int*   ei    = (const int*)d_in[1];
  const float* ew    = (const float*)d_in[2];
  const float* lin_w = (const float*)d_in[3];
  const float* lin_b = (const float*)d_in[4];
  const float* att_w = (const float*)d_in[5];
  const float* att_b = (const float*)d_in[6];
  const float* le1_w = (const float*)d_in[7];
  const float* le1_b = (const float*)d_in[8];
  const float* le2_w = (const float*)d_in[9];
  const float* le3_w = (const float*)d_in[10];
  const float* le3_b = (const float*)d_in[11];

  const int D  = in_sizes[4];            // 256
  const int N  = in_sizes[0] / D;        // 4096
  const long E0 = in_sizes[1] / 2;       // 131072
  const long Et = E0 + N;                // + self loops
  const int Kc = (N + 1) / 2;            // ceil(0.5*N) = 2048
  const int* srcI = ei;
  const int* dstI = ei + E0;

  // workspace carve (256B-aligned regions)
  char* w = (char*)d_ws;
  auto alloc = [&](size_t bytes) {
    char* p = w; w += (bytes + 255) & ~(size_t)255; return p;
  };
  float* xq     = (float*)alloc((size_t)N * D * 4);
  float* xn     = (float*)alloc((size_t)N * D * 4);
  float* smax   = (float*)alloc((size_t)N * 4);
  float* denomv = (float*)alloc((size_t)N * 4);
  float* qv     = (float*)alloc((size_t)N * 4);
  float* pv     = (float*)alloc((size_t)N * 4);
  float* av     = (float*)alloc((size_t)N * 4);
  float* bv     = (float*)alloc((size_t)N * 4);
  float* c3v    = (float*)alloc((size_t)N * 4);
  float* aggv   = (float*)alloc((size_t)N * 4);
  float* fitv   = (float*)alloc((size_t)N * 4);
  float* vvec   = (float*)alloc((size_t)D * 4);
  float* c0     = (float*)alloc(256);
  float* score  = (float*)alloc((size_t)Et * 4);
  float* alphav = (float*)alloc((size_t)Et * 4);
  int*   perm    = (int*)alloc((size_t)Kc * 4);
  int*   invperm = (int*)alloc((size_t)N * 4);
  float* Sp = (float*)alloc((size_t)N * Kc * 4);
  float* T  = (float*)alloc((size_t)N * Kc * 4);
  __bf16* SpT = (__bf16*)alloc((size_t)Kc * N * 2);  // Sp^T, K-major [Kc, N]
  __bf16* Tt  = (__bf16*)alloc((size_t)Kc * N * 2);  // T^T,  K-major [Kc, N]

  float* outX    = (float*)d_out;                      // [Kc, D]
  float* outA    = outX + (size_t)Kc * D;              // [Kc, Kc]
  float* outPerm = outA + (size_t)Kc * Kc;             // [Kc]

  const int TPB = 256;
  auto nb = [](long total, int tpb) {
    long b = (total + tpb - 1) / tpb;
    if (b > 1048576) b = 1048576;
    if (b < 1) b = 1;
    return (int)b;
  };
  const float NINF = -__builtin_inff();

  // init accumulators (every call; harness does not re-poison)
  k_fill_f32<<<nb((long)N * D, TPB), TPB, 0, stream>>>(xq, NINF, (long)N * D);
  k_fill_f32<<<nb((long)N * D, TPB), TPB, 0, stream>>>(xn, 0.f, (long)N * D);
  k_fill_f32<<<nb(N, TPB), TPB, 0, stream>>>(smax, NINF, N);
  k_fill_f32<<<nb(N, TPB), TPB, 0, stream>>>(denomv, 0.f, N);
  k_fill_f32<<<nb(N, TPB), TPB, 0, stream>>>(aggv, 0.f, N);
  k_fill_i32<<<nb(N, TPB), TPB, 0, stream>>>(invperm, -1, N);
  k_fill_f32<<<nb((long)N * Kc, TPB), TPB, 0, stream>>>(Sp, 0.f, (long)N * Kc);
  k_fill_f32<<<nb((long)N * Kc, TPB), TPB, 0, stream>>>(T, 0.f, (long)N * Kc);

  // attention path (x_q @ lin_w collapsed to matvec against lin_w @ att_w[:D])
  k_vvec<<<1, TPB, 0, stream>>>(lin_w, lin_b, att_w, att_b, vvec, c0, D);
  k_xq_max<<<nb(Et * D, TPB), TPB, 0, stream>>>(x, srcI, dstI, xq, E0, Et, D);
  k_node_qp<<<nb((long)N * 32, TPB), TPB, 0, stream>>>(xq, x, vvec, c0, att_w, qv, pv, N, D);
  k_score1<<<nb(Et, TPB), TPB, 0, stream>>>(srcI, dstI, qv, pv, score, smax, E0, Et);
  k_score2<<<nb(Et, TPB), TPB, 0, stream>>>(dstI, score, smax, denomv, E0, Et);
  k_alpha<<<nb(Et, TPB), TPB, 0, stream>>>(dstI, score, denomv, alphav, E0, Et);
  k_xn<<<nb(Et * D, TPB), TPB, 0, stream>>>(srcI, dstI, alphav, x, xn, E0, Et, D);

  // LEConv fitness
  k_le<<<nb((long)N * 32, TPB), TPB, 0, stream>>>(xn, le1_w, le1_b, le2_w, le3_w,
                                                  av, bv, c3v, N, D);
  k_agg<<<nb(Et, TPB), TPB, 0, stream>>>(srcI, dstI, av, bv, aggv, E0, Et);
  k_fit<<<nb(N, TPB), TPB, 0, stream>>>(aggv, c3v, le3_b, fitv, N);

  // exact top-k (O(N^2) rank) + pooled features
  k_topk<<<nb(N, TPB), TPB, 0, stream>>>(fitv, perm, invperm, outPerm, N, Kc);
  k_xout<<<nb((long)Kc * D, TPB), TPB, 0, stream>>>(perm, fitv, xn, outX, Kc, D);

  // coarsening: Sp dense from sparse S; T = A @ Sp via edge scatter (A is sparse)
  k_sp<<<nb(Et, TPB), TPB, 0, stream>>>(srcI, dstI, invperm, alphav, Sp, E0, Et, Kc);
  k_tscatter<<<nb(Et * Kc, TPB), TPB, 0, stream>>>(srcI, dstI, ew, Sp, T, E0, Et, Kc);

  // A_c = Sp^T @ T : dense 2048x2048x4096 bf16 WMMA GEMM, f32 accumulate
  k_transpose_bf16<<<nb((long)Kc * N, TPB), TPB, 0, stream>>>(Sp, SpT, N, Kc);
  k_transpose_bf16<<<nb((long)Kc * N, TPB), TPB, 0, stream>>>(T, Tt, N, Kc);
  int totalBlocks = (Kc / 32) * (Kc / 32);
  k_gemm_bf16_2x2<<<nb((long)totalBlocks * 32, TPB), TPB, 0, stream>>>(
      SpT, Tt, outA, Kc, Kc, N, Kc, totalBlocks);
  k_diag0<<<nb(Kc, TPB), TPB, 0, stream>>>(outA, Kc);
}
